// CrossModalAttention_1769526526021
// MI455X (gfx1250) — matrix operand order repaired and checked by reading, not compiled, not run on hardware
//
#include <hip/hip_runtime.h>
#include <hip/hip_bf16.h>

// ---------------------------------------------------------------------------
// MI455X (gfx1250) fused cross-modal attention.
//  - All GEMMs on v_wmma_f32_16x16x32_f16 (fp32 accumulate).
//  - Double-buffered LDS-staged GEMM tiles (1 barrier / k-step).
//  - Async global->LDS copies (GLOBAL_LOAD_ASYNC_TO_LDS_*) for raw f16 tiles.
//  - Fused softmax + attn-mean output per (batch, 16-row) tile.
// ---------------------------------------------------------------------------

typedef __attribute__((ext_vector_type(16))) _Float16 v16h;
typedef __attribute__((ext_vector_type(8)))  _Float16 v8h;
typedef __attribute__((ext_vector_type(8)))  float    v8f;

#define DDIM  768
#define BATCH 8
#define LDIM  512
#define NDIM  1024
#define NHEAD 12
#define HD    64

static __device__ __forceinline__ v16h cat16(v8h lo, v8h hi) {
  v16h r;
#pragma unroll
  for (int i = 0; i < 8; ++i) { r[i] = lo[i]; r[i + 8] = hi[i]; }
  return r;
}

static __device__ __forceinline__ v8f wmma16(v16h a, v16h b, v8f c) {
  // (neg_a, A, neg_b, B, c_mod, C, reuse_a, reuse_b)
  return __builtin_amdgcn_wmma_f32_16x16x32_f16(false, a, false, b, (short)0, c,
                                                false, false);
}

static __device__ __forceinline__ v8f vzero8() {
  v8f c;
#pragma unroll
  for (int i = 0; i < 8; ++i) c[i] = 0.0f;
  return c;
}

// ---- CDNA5 async global->LDS copies (ISA 10.x / 15.18.3, ASYNCcnt-tracked).
// LDS dest offset = low 32 bits of the flat shared-memory address.
static __device__ __forceinline__ void async_copy_b128(void* lds, const void* g) {
  asm volatile("global_load_async_to_lds_b128 %0, %1, off"
               :
               : "v"((unsigned)(uintptr_t)lds),
                 "v"((unsigned long long)(uintptr_t)g)
               : "memory");
}
static __device__ __forceinline__ void async_copy_b64(void* lds, const void* g) {
  asm volatile("global_load_async_to_lds_b64 %0, %1, off"
               :
               : "v"((unsigned)(uintptr_t)lds),
                 "v"((unsigned long long)(uintptr_t)g)
               : "memory");
}
static __device__ __forceinline__ void wait_async0() {
  asm volatile("s_wait_asynccnt 0x0" ::: "memory");
}

// ---------------------------------------------------------------------------
// Generic 768-K GEMM: C[M x 768] = A[M x 768] * W[768 x 768] + bias.
//  A_F16:    A is f16 (raw async-copy stage-in), else fp32 (convert on stage).
//  OUT_F16:  store f16 (else fp32).
//  STORE_T:  store transposed per-batch:  C[(b*768+col)*1024 + row%1024]
//            (used for V so attention B-fragments are contiguous).
// 256 thr = 8 waves, 64x64 tile, each wave a 16x32 sub-tile (2 WMMA/k-step),
// K-loop = 24 steps of 32, double-buffered LDS.
// ---------------------------------------------------------------------------
template <bool A_F16, bool OUT_F16, bool STORE_T>
__global__ __launch_bounds__(256) void gemm768(const void* __restrict__ Ap,
                                               const float* __restrict__ W,
                                               const float* __restrict__ bias,
                                               void* __restrict__ Cp) {
  __shared__ __align__(16) _Float16 As[2][64][40];   // [m][k], stride 80B
  __shared__ __align__(16) _Float16 Bs[2][64][40];   // [n][k] (W transposed)

  const int tid = threadIdx.x;
  const int m0 = blockIdx.x * 64, n0 = blockIdx.y * 64;
  const int w = tid >> 5, lane = tid & 31, ln = lane & 15, kh = lane >> 4;
  const int mi = w & 3, nj = w >> 2;                 // 4 m-tiles x 2 n-pairs
  const int ar = tid >> 2, aseg = (tid & 3) * 8;     // A stage mapping
  const int bk = tid >> 3, bn8 = (tid & 7) * 8;      // W stage mapping

  v8f c0 = vzero8(), c1 = vzero8();

  // ---- staging lambda: fill buffer `buf` with k-chunk starting at k0
  auto stage = [&](int buf, int k0) {
    if (A_F16) {
      const _Float16* Ah = (const _Float16*)Ap + (size_t)(m0 + ar) * DDIM + k0 + aseg;
      async_copy_b128(&As[buf][ar][aseg], Ah);
    } else {
      const float* Af = (const float*)Ap + (size_t)(m0 + ar) * DDIM + k0 + aseg;
      v8h v;
#pragma unroll
      for (int i = 0; i < 8; ++i) v[i] = (_Float16)Af[i];
      *(v8h*)&As[buf][ar][aseg] = v;
    }
    const float* src = W + (size_t)(k0 + bk) * DDIM + n0 + bn8;
#pragma unroll
    for (int i = 0; i < 8; ++i) Bs[buf][bn8 + i][bk] = (_Float16)src[i];
  };

  stage(0, 0);
  if (A_F16) wait_async0();
  __syncthreads();

  for (int kt = 0; kt < 24; ++kt) {
    const int cur = kt & 1;
    if (kt < 23) stage(cur ^ 1, (kt + 1) * 32);

    // ---- WMMA fragments: A and B both use contiguous k = 16*kh + s
    const int am = mi * 16 + ln;
    v16h a = cat16(*(const v8h*)&As[cur][am][kh * 16],
                   *(const v8h*)&As[cur][am][kh * 16 + 8]);
    const int bn0r = (2 * nj + 0) * 16 + ln;
    const int bn1r = (2 * nj + 1) * 16 + ln;
    v16h b0 = cat16(*(const v8h*)&Bs[cur][bn0r][kh * 16],
                    *(const v8h*)&Bs[cur][bn0r][kh * 16 + 8]);
    v16h b1 = cat16(*(const v8h*)&Bs[cur][bn1r][kh * 16],
                    *(const v8h*)&Bs[cur][bn1r][kh * 16 + 8]);
    c0 = wmma16(a, b0, c0);
    c1 = wmma16(a, b1, c1);

    if (A_F16 && kt < 23) wait_async0();
    __syncthreads();
  }

  // ---- epilogue: C layout VGPR r -> row (r + 8*kh), col = lane&15
#pragma unroll
  for (int r = 0; r < 8; ++r) {
    const int mrow = m0 + mi * 16 + r + 8 * kh;
#pragma unroll
    for (int s = 0; s < 2; ++s) {
      const int ncol = n0 + (2 * nj + s) * 16 + ln;
      const float val = (s ? c1[r] : c0[r]) + bias[ncol];
      if (OUT_F16) {
        _Float16* C = (_Float16*)Cp;
        if (STORE_T) {
          const int bi = mrow >> 10, nr = mrow & 1023;
          C[((size_t)(bi * DDIM + ncol)) * NDIM + nr] = (_Float16)val;
        } else {
          C[(size_t)mrow * DDIM + ncol] = (_Float16)val;
        }
      } else {
        ((float*)Cp)[(size_t)mrow * DDIM + ncol] = val;
      }
    }
  }
}

// ---------------------------------------------------------------------------
// Fused attention: one workgroup per (batch, 16 query rows).
// Loops over 12 heads. Per head:
//   scores S(16x1024) = Q(16x64) K^T, split: wave w owns columns [w*128, +128)
//   softmax (shfl + LDS cross-wave reduction), attn-mean RMW to d_out,
//   attended(16x64) = P(16x1024) @ V, K-dim split across waves, LDS atomic
//   reduction, f16 store to workspace for the final O projection.
// ---------------------------------------------------------------------------
__global__ __launch_bounds__(256) void attn_fused(const void* __restrict__ Qp,
                                                  const void* __restrict__ Kp,
                                                  const void* __restrict__ Vp,
                                                  void* __restrict__ Op,
                                                  float* __restrict__ attn_mean) {
  const _Float16* Qh = (const _Float16*)Qp;
  const _Float16* Kh = (const _Float16*)Kp;
  const _Float16* Vt = (const _Float16*)Vp;   // [b][col(768)][n(1024)]
  _Float16* Ao = (_Float16*)Op;

  __shared__ __align__(16) _Float16 Qs[16][72];        // 2.25 KB
  __shared__ __align__(16) _Float16 Ps[8][16][128];    // 32 KB (per-wave A tiles)
  __shared__ float red[16][8];                         // row reductions
  __shared__ float Af[16][64];                         // attended accumulator

  const int tid = threadIdx.x;
  const int w = tid >> 5, lane = tid & 31, ln = lane & 15, kh = lane >> 4;
  const int b = blockIdx.x >> 5, l0 = (blockIdx.x & 31) * 16;
  const float scale = 0.125f;                // d^-0.5, d=64
  const float invH = 1.0f / (float)NHEAD;

  for (int h = 0; h < NHEAD; ++h) {
    // ---- async-load Q tile (16 x 64 f16 = 2KB) for this head into LDS
    {
      const int row = tid >> 4, c4 = (tid & 15) * 4;
      async_copy_b64(&Qs[row][c4],
                     Qh + ((size_t)(b * LDIM + l0 + row)) * DDIM + h * HD + c4);
      wait_async0();
    }
    __syncthreads();

    // ---- scores: 8 sub-tiles of 16 cols per wave, K fragments from global
    v8f sc[8];
#pragma unroll
    for (int s = 0; s < 8; ++s) sc[s] = vzero8();
#pragma unroll
    for (int kc = 0; kc < 2; ++kc) {
      v16h a = cat16(*(const v8h*)&Qs[ln][kc * 32 + kh * 16],
                     *(const v8h*)&Qs[ln][kc * 32 + kh * 16 + 8]);
#pragma unroll
      for (int s = 0; s < 8; ++s) {
        const int ng = w * 128 + s * 16 + ln;
        const _Float16* kp =
            Kh + ((size_t)(b * NDIM + ng)) * DDIM + h * HD + kc * 32 + kh * 16;
        sc[s] = wmma16(a, cat16(*(const v8h*)kp, *(const v8h*)(kp + 8)), sc[s]);
      }
    }

    // ---- softmax: row max
    float pmax[8];
#pragma unroll
    for (int r = 0; r < 8; ++r) pmax[r] = -3.0e38f;
#pragma unroll
    for (int s = 0; s < 8; ++s)
#pragma unroll
      for (int r = 0; r < 8; ++r) {
        sc[s][r] *= scale;
        pmax[r] = fmaxf(pmax[r], sc[s][r]);
      }
#pragma unroll
    for (int off = 1; off < 16; off <<= 1)
#pragma unroll
      for (int r = 0; r < 8; ++r)
        pmax[r] = fmaxf(pmax[r], __shfl_xor(pmax[r], off, 32));
    if (ln == 0)
#pragma unroll
      for (int r = 0; r < 8; ++r) red[r + 8 * kh][w] = pmax[r];
    __syncthreads();
    float gmax[8];
#pragma unroll
    for (int r = 0; r < 8; ++r) {
      float m = red[r + 8 * kh][0];
#pragma unroll
      for (int ww = 1; ww < 8; ++ww) m = fmaxf(m, red[r + 8 * kh][ww]);
      gmax[r] = m;
    }
    __syncthreads();

    // ---- exp + row sum
    float psum[8];
#pragma unroll
    for (int r = 0; r < 8; ++r) psum[r] = 0.0f;
#pragma unroll
    for (int s = 0; s < 8; ++s)
#pragma unroll
      for (int r = 0; r < 8; ++r) {
        const float p = __expf(sc[s][r] - gmax[r]);
        sc[s][r] = p;
        psum[r] += p;
      }
#pragma unroll
    for (int off = 1; off < 16; off <<= 1)
#pragma unroll
      for (int r = 0; r < 8; ++r) psum[r] += __shfl_xor(psum[r], off, 32);
    if (ln == 0)
#pragma unroll
      for (int r = 0; r < 8; ++r) red[r + 8 * kh][w] = psum[r];
    __syncthreads();
    float ginv[8];
#pragma unroll
    for (int r = 0; r < 8; ++r) {
      float su = 0.0f;
#pragma unroll
      for (int ww = 0; ww < 8; ++ww) su += red[r + 8 * kh][ww];
      ginv[r] = 1.0f / su;
    }
    __syncthreads();

    // ---- finalize P, emit attn-mean (owned per-thread across h loop),
    //      stash P as f16 A-tiles in per-wave LDS
#pragma unroll
    for (int s = 0; s < 8; ++s)
#pragma unroll
      for (int r = 0; r < 8; ++r) {
        const float p = sc[s][r] * ginv[r];
        sc[s][r] = p;
        const int mrow = l0 + r + 8 * kh;
        const int ng = w * 128 + s * 16 + ln;
        float* dst = attn_mean + ((size_t)(b * LDIM + mrow)) * NDIM + ng;
        if (h == 0) *dst = p * invH;
        else        *dst += p * invH;
        Ps[w][r + 8 * kh][s * 16 + ln] = (_Float16)p;
      }
    // clear attended accumulator
#pragma unroll
    for (int i = 0; i < 4; ++i) ((float*)Af)[tid * 4 + i] = 0.0f;
    __syncthreads();

    // ---- attended = P @ V_h ; wave w owns K-chunk [w*128, +128)
    v8f ac[4];
#pragma unroll
    for (int s = 0; s < 4; ++s) ac[s] = vzero8();
#pragma unroll
    for (int ks = 0; ks < 4; ++ks) {
      v16h a = cat16(*(const v8h*)&Ps[w][ln][ks * 32 + kh * 16],
                     *(const v8h*)&Ps[w][ln][ks * 32 + kh * 16 + 8]);
#pragma unroll
      for (int s = 0; s < 4; ++s) {
        const _Float16* vp = Vt + ((size_t)(b * DDIM + h * HD + s * 16 + ln)) * NDIM +
                             w * 128 + ks * 32 + kh * 16;
        ac[s] = wmma16(a, cat16(*(const v8h*)vp, *(const v8h*)(vp + 8)), ac[s]);
      }
    }
#pragma unroll
    for (int s = 0; s < 4; ++s)
#pragma unroll
      for (int r = 0; r < 8; ++r)
        atomicAdd(&Af[r + 8 * kh][s * 16 + ln], ac[s][r]);
    __syncthreads();

    // ---- store attended tile (f16) into workspace at column block h*64
    {
      const int e0 = tid * 4;
#pragma unroll
      for (int i = 0; i < 4; ++i) {
        const int e = e0 + i, row = e >> 6, col = e & 63;
        Ao[((size_t)(b * LDIM + l0 + row)) * DDIM + h * HD + col] =
            (_Float16)Af[row][col];
      }
    }
    __syncthreads();   // protect Qs/Ps/Af before next head
  }
}

// ---------------------------------------------------------------------------
// Host-side launcher. Workspace layout (f16):
//   Q  : 4096*768            ( 6.29 MB)
//   K  : 8192*768            (12.58 MB)
//   Vt : 8*768*1024          (12.58 MB, transposed)
//   At : 4096*768            ( 6.29 MB)
// ---------------------------------------------------------------------------
extern "C" void kernel_launch(void* const* d_in, const int* in_sizes, int n_in,
                              void* d_out, int out_size, void* d_ws, size_t ws_size,
                              hipStream_t stream) {
  (void)in_sizes; (void)n_in; (void)out_size; (void)ws_size;
  const float* tokens  = (const float*)d_in[0];
  const float* patches = (const float*)d_in[1];
  const float* Wq = (const float*)d_in[2]; const float* bq = (const float*)d_in[3];
  const float* Wk = (const float*)d_in[4]; const float* bk = (const float*)d_in[5];
  const float* Wv = (const float*)d_in[6]; const float* bv = (const float*)d_in[7];
  const float* Wo = (const float*)d_in[8]; const float* bo = (const float*)d_in[9];

  float* out = (float*)d_out;                           // (8,512,768)
  float* attn_mean = out + (size_t)BATCH * LDIM * DDIM; // (8,512,1024)

  char* ws = (char*)d_ws;
  void* wsQ  = ws;
  void* wsK  = ws + (size_t)6291456;
  void* wsVt = ws + (size_t)6291456 + 12582912;
  void* wsAt = ws + (size_t)6291456 + 12582912 + 12582912;

  dim3 blk(256);
  // Q = tokens @ Wq + bq           (4096 x 768)
  gemm768<false, true, false><<<dim3(64, 12), blk, 0, stream>>>(tokens, Wq, bq, wsQ);
  // K = patches @ Wk + bk          (8192 x 768)
  gemm768<false, true, false><<<dim3(128, 12), blk, 0, stream>>>(patches, Wk, bk, wsK);
  // V = patches @ Wv + bv, stored transposed per batch (768 x 1024)
  gemm768<false, true, true><<<dim3(128, 12), blk, 0, stream>>>(patches, Wv, bv, wsVt);
  // fused attention + attn-mean
  attn_fused<<<dim3(BATCH * (LDIM / 16)), blk, 0, stream>>>(wsQ, wsK, wsVt, wsAt,
                                                            attn_mean);
  // out = attended @ Wo + bo       (4096 x 768, fp32 out)
  gemm768<true, false, false><<<dim3(64, 12), blk, 0, stream>>>(wsAt, Wo, bo, out);
}
